// DescrptDPA2_25984552141223
// MI455X (gfx1250) — matrix operand order repaired and checked by reading, not compiled, not run on hardware
//
#include <hip/hip_runtime.h>
#include <cstdint>
#include <cstddef>

// ---------------------------------------------------------------------------
// DPA-2 descriptor forward for MI455X (gfx1250, wave32, WMMA).
// All dense projections -> one generic WMMA GEMM (f16 frags, f32 accum,
// v_wmma_f32_16x16x32_f16).  M%32==0 and N%16==0 hold at every call site,
// so the hot loop has NO predication.  Each wave computes a 32x(16*NT)
// macro-tile (MT=2 x NT tiles), reusing B fragments across both M tiles
// to cut vmem instructions per WMMA.  K tails (K=16, K=720) use address
// clamping + value selects, never exec-mask branches.
// ---------------------------------------------------------------------------

typedef __attribute__((ext_vector_type(16))) _Float16 v16h;
typedef __attribute__((ext_vector_type(8)))  float    v8f;

#define NLOC     4096
#define NALL     5120
#define RI_NSEL  120
#define RF_NSEL  40
#define TEBD     8
#define G1C      128
#define G2C      16
#define NH1      4
#define HD1      64
#define NH2      4
#define NLAYERS  3
#define DIN1     720                 // G1 + G2 + G2*4 + G1*4
#define RROWS    (NLOC * RF_NSEL)    // 163840

// ---------------------------------------------------------------------------
// GEMM: C[M,N] = act(A[M,K] @ B[K,N] + bias).
// Fragment layouts per CDNA5 ISA 7.12.2 (wave32):
//   A 16x32 f16 : lane L -> row M = L&15; elems 0..7  -> K = k0+(L>>4)*8+e
//                                         elems 8..15 -> K = k0+16+(L>>4)*8+e
//   B 32x16 f16 : lane L -> col N = L&15; elem e -> K = k0+(L>>4)*8+e+(e&8) (same order as A)
//   C/D 16x16 f32: VGPR r, lanes 0-15 -> M=r, lanes 16-31 -> M=8+r; N = L&15
// ---------------------------------------------------------------------------
template<int ACT, int NT, int MT>
__global__ __launch_bounds__(128)
void k_gemm(const float* __restrict__ A, const float* __restrict__ B,
            const float* __restrict__ bias, float* __restrict__ C,
            int M, int N, int K, int totTiles)
{
    const int lane = threadIdx.x & 31;
    const int wave = threadIdx.x >> 5;
    const int tid  = blockIdx.x * 4 + wave;
    if (tid >= totTiles) return;                    // wave-uniform

    const unsigned ntg = (unsigned)(N >> 4) / (unsigned)NT;
    const int tmg = (int)((unsigned)tid / ntg);
    const int tng = (int)((unsigned)tid % ntg);
    const int half = lane >> 4;
    const int rowb = tmg * (16 * MT) + (lane & 15);
    const int colb = tng * (16 * NT) + (lane & 15);

    const float* __restrict__ Arow[MT];
#pragma unroll
    for (int mt = 0; mt < MT; ++mt) Arow[mt] = A + (size_t)(rowb + 16 * mt) * K;

    v8f acc[MT][NT];
#pragma unroll
    for (int mt = 0; mt < MT; ++mt)
#pragma unroll
        for (int nt = 0; nt < NT; ++nt) acc[mt][nt] = (v8f){};

    const int Kfull = K & ~31;

    for (int k0 = 0; k0 < Kfull; k0 += 32) {
        // ---- A fragments: two contiguous 8-float runs -> 4x b128 each ----
        v16h af[MT];
#pragma unroll
        for (int mt = 0; mt < MT; ++mt) {
            const float* ap = Arow[mt] + k0 + half * 8;
            float4 a0 = *(const float4*)(ap);
            float4 a1 = *(const float4*)(ap + 4);
            float4 a2 = *(const float4*)(ap + 16);
            float4 a3 = *(const float4*)(ap + 20);
            af[mt][0]  = (_Float16)a0.x; af[mt][1]  = (_Float16)a0.y;
            af[mt][2]  = (_Float16)a0.z; af[mt][3]  = (_Float16)a0.w;
            af[mt][4]  = (_Float16)a1.x; af[mt][5]  = (_Float16)a1.y;
            af[mt][6]  = (_Float16)a1.z; af[mt][7]  = (_Float16)a1.w;
            af[mt][8]  = (_Float16)a2.x; af[mt][9]  = (_Float16)a2.y;
            af[mt][10] = (_Float16)a2.z; af[mt][11] = (_Float16)a2.w;
            af[mt][12] = (_Float16)a3.x; af[mt][13] = (_Float16)a3.y;
            af[mt][14] = (_Float16)a3.z; af[mt][15] = (_Float16)a3.w;
        }
        // ---- B fragments: shared by both M tiles ----
        const float* bp = B + (size_t)(k0 + half * 8) * N + colb;
        v16h bf[NT];
#pragma unroll
        for (int nt = 0; nt < NT; ++nt)
#pragma unroll
            for (int e = 0; e < 16; ++e)
                bf[nt][e] = (_Float16)bp[(size_t)(e + (e & 8)) * N + 16 * nt];
#pragma unroll
        for (int mt = 0; mt < MT; ++mt)
#pragma unroll
            for (int nt = 0; nt < NT; ++nt)
                acc[mt][nt] = __builtin_amdgcn_wmma_f32_16x16x32_f16(
                    false, af[mt], false, bf[nt], (short)0, acc[mt][nt], false, false);
    }

    if (Kfull < K) {   // K tail: clamp addresses, select values (no branches)
        const int k0 = Kfull;
        v16h af[MT];
#pragma unroll
        for (int mt = 0; mt < MT; ++mt)
#pragma unroll
            for (int e = 0; e < 8; ++e) {
                int ka = k0 + half * 8 + e;
                int kb = ka + 16;
                float va = Arow[mt][(ka < K) ? ka : 0];
                float vb = Arow[mt][(kb < K) ? kb : 0];
                if (ka >= K) va = 0.0f;
                if (kb >= K) vb = 0.0f;
                af[mt][e]     = (_Float16)va;
                af[mt][e + 8] = (_Float16)vb;
            }
        v16h bf[NT];
#pragma unroll
        for (int e = 0; e < 16; ++e) {
            int kk = k0 + half * 8 + e + (e & 8);
            const float* brow = B + (size_t)((kk < K) ? kk : 0) * N + colb;
#pragma unroll
            for (int nt = 0; nt < NT; ++nt) {
                float bv = brow[16 * nt];
                if (kk >= K) bv = 0.0f;
                bf[nt][e] = (_Float16)bv;
            }
        }
#pragma unroll
        for (int mt = 0; mt < MT; ++mt)
#pragma unroll
            for (int nt = 0; nt < NT; ++nt)
                acc[mt][nt] = __builtin_amdgcn_wmma_f32_16x16x32_f16(
                    false, af[mt], false, bf[nt], (short)0, acc[mt][nt], false, false);
    }

    // ---- epilogue: unguarded (M%32==0, N%16==0) ----
#pragma unroll
    for (int nt = 0; nt < NT; ++nt) {
        const int col = colb + 16 * nt;
        const float badd = bias ? bias[col] : 0.0f;
#pragma unroll
        for (int mt = 0; mt < MT; ++mt) {
            float* crow = C + (size_t)(tmg * (16 * MT) + 16 * mt + half * 8) * N + col;
#pragma unroll
            for (int r = 0; r < 8; ++r) {
                float v = acc[mt][nt][r] + badd;
                if (ACT) v = tanhf(v);
                crow[(size_t)r * N] = v;
            }
        }
    }
}

static inline void gemm(const float* A, const float* B, const float* bias,
                        float* C, int M, int N, int K, int act, hipStream_t s)
{
    const int mt = M >> 5;             // M % 32 == 0 at every call site (MT=2)
    const int nt = N >> 4;
    if ((nt & 1) == 0) {
        const int tiles = mt * (nt >> 1);
        dim3 grid((tiles + 3) >> 2);
        if (act) k_gemm<1, 2, 2><<<grid, 128, 0, s>>>(A, B, bias, C, M, N, K, tiles);
        else     k_gemm<0, 2, 2><<<grid, 128, 0, s>>>(A, B, bias, C, M, N, K, tiles);
    } else {
        const int tiles = mt * nt;
        dim3 grid((tiles + 3) >> 2);
        if (act) k_gemm<1, 1, 2><<<grid, 128, 0, s>>>(A, B, bias, C, M, N, K, tiles);
        else     k_gemm<0, 1, 2><<<grid, 128, 0, s>>>(A, B, bias, C, M, N, K, tiles);
    }
}

// ---------------------------------------------------------------------------
// tebd_ext[a][c] = tebd[atype_ext[a]][c]
// ---------------------------------------------------------------------------
__global__ void k_tebd(const int* __restrict__ atype, const float* __restrict__ tebd,
                       float* __restrict__ tebd_ext)
{
    int idx = blockIdx.x * blockDim.x + threadIdx.x;
    if (idx >= NALL * TEBD) return;
    int a = idx / TEBD, c = idx % TEBD;
    tebd_ext[idx] = tebd[atype[a] * TEBD + c];
}

// ---------------------------------------------------------------------------
// Repinit: per-atom env matrix + tiny embed MLP (fully unrolled into regs)
// + the two small einsum contractions; writes res[l][g*16+a] (g<100, a<16).
// ---------------------------------------------------------------------------
__global__ __launch_bounds__(128)
void k_repinit(const float* __restrict__ coord, const int* __restrict__ nlist,
               const float* __restrict__ tebd_ext,
               const float* __restrict__ W1, const float* __restrict__ B1,
               const float* __restrict__ W2, const float* __restrict__ B2,
               const float* __restrict__ W3, const float* __restrict__ B3,
               float* __restrict__ resbuf)
{
    __shared__ float s_gg[RI_NSEL][100];
    __shared__ float s_env[RI_NSEL][4];
    __shared__ float s_xyz[4][100];
    const int l = blockIdx.x;
    const int t = threadIdx.x;

    if (t < RI_NSEL) {
        const int j = nlist[l * RI_NSEL + t];
        float dx = coord[j * 3 + 0] - coord[l * 3 + 0];
        float dy = coord[j * 3 + 1] - coord[l * 3 + 1];
        float dz = coord[j * 3 + 2] - coord[l * 3 + 2];
        float r2 = fmaf(dx, dx, fmaf(dy, dy, fmaf(dz, dz, 1e-12f)));
        float r  = sqrtf(r2);
        float uu = r - 8.0f;                              // (rc-rs) == 1
        float u2 = uu * uu;
        float poly = uu * u2 * fmaf(-6.0f, u2, fmaf(15.0f, uu, -10.0f)) + 1.0f;
        float sw = (r < 8.0f) ? 1.0f : ((r < 9.0f) ? poly : 0.0f);
        float e0 = sw / r;
        float sc = sw / r2;
        s_env[t][0] = e0;
        s_env[t][1] = dx * sc;
        s_env[t][2] = dy * sc;
        s_env[t][3] = dz * sc;

        float x[17];
        x[0] = e0;
#pragma unroll
        for (int c = 0; c < TEBD; ++c) {
            x[1 + c] = tebd_ext[l * TEBD + c];
            x[9 + c] = tebd_ext[j * TEBD + c];
        }
        float h1[25];
#pragma unroll
        for (int o = 0; o < 25; ++o) {
            float acc = B1[o];
#pragma unroll
            for (int i = 0; i < 17; ++i) acc = fmaf(x[i], W1[i * 25 + o], acc);
            h1[o] = tanhf(acc);
        }
        float h2[50];
#pragma unroll
        for (int o = 0; o < 50; ++o) {
            float acc = B2[o];
#pragma unroll
            for (int i = 0; i < 25; ++i) acc = fmaf(h1[i], W2[i * 50 + o], acc);
            h2[o] = tanhf(acc) + h1[(o < 25) ? o : o - 25];   // skip: h + [x,x]
        }
#pragma unroll
        for (int o = 0; o < 100; ++o) {
            float acc = B3[o];
#pragma unroll
            for (int i = 0; i < 50; ++i) acc = fmaf(h2[i], W3[i * 100 + o], acc);
            s_gg[t][o] = tanhf(acc) + h2[(o < 50) ? o : o - 50];
        }
    }
    __syncthreads();
    // xyz[i][g] = sum_n env[n][i] * gg[n][g] / 120
    for (int idx = t; idx < 400; idx += 128) {
        int i = idx / 100, g = idx % 100;
        float acc = 0.0f;
        for (int n = 0; n < RI_NSEL; ++n) acc = fmaf(s_env[n][i], s_gg[n][g], acc);
        s_xyz[i][g] = acc * (1.0f / RI_NSEL);
    }
    __syncthreads();
    // res[g][a] = sum_i xyz[i][g]*xyz[i][a]  (a < 16)
    for (int idx = t; idx < 1600; idx += 128) {
        int g = idx >> 4, a = idx & 15;
        float acc = 0.0f;
#pragma unroll
        for (int i = 0; i < 4; ++i) acc = fmaf(s_xyz[i][g], s_xyz[i][a], acc);
        resbuf[(size_t)l * 1600 + idx] = acc;
    }
}

// ---------------------------------------------------------------------------
// Repformer init: env2 -> sw2, h2, g2 = t0 * g2w + g2b   (one thread / pair)
// ---------------------------------------------------------------------------
__global__ void k_rf_init(const float* __restrict__ coord, const int* __restrict__ nlist,
                          const float* __restrict__ g2w, const float* __restrict__ g2b,
                          float* __restrict__ g2, float* __restrict__ h2,
                          float* __restrict__ sw2)
{
    int idx = blockIdx.x * blockDim.x + threadIdx.x;
    if (idx >= RROWS) return;
    int l = idx / RF_NSEL;
    int j = nlist[idx];
    float dx = coord[j * 3 + 0] - coord[l * 3 + 0];
    float dy = coord[j * 3 + 1] - coord[l * 3 + 1];
    float dz = coord[j * 3 + 2] - coord[l * 3 + 2];
    float r2 = fmaf(dx, dx, fmaf(dy, dy, fmaf(dz, dz, 1e-12f)));
    float r  = sqrtf(r2);
    float uu = (r - 3.5f) * 2.0f;                          // /(rc-rs)=0.5
    float u2 = uu * uu;
    float poly = uu * u2 * fmaf(-6.0f, u2, fmaf(15.0f, uu, -10.0f)) + 1.0f;
    float sw = (r < 3.5f) ? 1.0f : ((r < 4.0f) ? poly : 0.0f);
    sw2[idx] = sw;
    float t0 = sw / r;
    float sc = sw / r2;
    h2[idx * 3 + 0] = dx * sc;
    h2[idx * 3 + 1] = dy * sc;
    h2[idx * 3 + 2] = dz * sc;
#pragma unroll
    for (int g = 0; g < G2C; ++g)
        g2[(size_t)idx * G2C + g] = fmaf(t0, g2w[g], g2b[g]);
}

// gg1[row][c] = g1[mapping[nlist[row]]][c];  g1gg1 = g1[l] (.) gg1
__global__ void k_gather(const float* __restrict__ g1, const int* __restrict__ nlist,
                         const int* __restrict__ mapping,
                         float* __restrict__ gg1, float* __restrict__ g1gg1)
{
    size_t idx = (size_t)blockIdx.x * blockDim.x + threadIdx.x;
    if (idx >= (size_t)RROWS * G1C) return;
    int row = (int)(idx >> 7), c = (int)(idx & 127);
    int l = row / RF_NSEL;
    int m = mapping[nlist[row]];
    float v = g1[(size_t)m * G1C + c];
    gg1[idx] = v;
    g1gg1[idx] = v * g1[(size_t)l * G1C + c];
}

// g2 neighbor attention: block = (atom, head); 40-wide softmax per row.
// All m-loops fully unrolled so logit[] stays in VGPRs (no scratch).
__global__ __launch_bounds__(64)
void k_attn2(const float* __restrict__ q2, const float* __restrict__ k2,
             const float* __restrict__ v2, const float* __restrict__ sw2,
             float* __restrict__ o2)
{
    __shared__ float s_q[RF_NSEL][G2C], s_k[RF_NSEL][G2C], s_v[RF_NSEL][G2C];
    __shared__ float s_sw[RF_NSEL];
    const int l = blockIdx.x, h = blockIdx.y, t = threadIdx.x;
    const size_t base = (size_t)l * RF_NSEL;
    if (t < RF_NSEL) {
        const size_t r = (base + t) * (NH2 * G2C) + h * G2C;
#pragma unroll
        for (int d = 0; d < G2C; ++d) {
            s_q[t][d] = q2[r + d]; s_k[t][d] = k2[r + d]; s_v[t][d] = v2[r + d];
        }
        s_sw[t] = sw2[base + t];
    }
    __syncthreads();
    if (t < RF_NSEL) {
        float logit[RF_NSEL];
        float mx = -3.0e38f;
#pragma unroll
        for (int m = 0; m < RF_NSEL; ++m) {
            float acc = 0.0f;
#pragma unroll
            for (int d = 0; d < G2C; ++d) acc = fmaf(s_q[t][d], s_k[m][d], acc);
            acc *= 0.25f;                     // 1/sqrt(16)
            logit[m] = acc;
            mx = fmaxf(mx, acc);
        }
        float sum = 0.0f;
#pragma unroll
        for (int m = 0; m < RF_NSEL; ++m) { float e = __expf(logit[m] - mx); logit[m] = e; sum += e; }
        float inv = s_sw[t] / sum;
        float o[G2C];
#pragma unroll
        for (int d = 0; d < G2C; ++d) o[d] = 0.0f;
#pragma unroll
        for (int m = 0; m < RF_NSEL; ++m) {
            float w = logit[m] * inv * s_sw[m];
#pragma unroll
            for (int d = 0; d < G2C; ++d) o[d] = fmaf(w, s_v[m][d], o[d]);
        }
        const size_t r = (base + t) * (NH2 * G2C) + h * G2C;
#pragma unroll
        for (int d = 0; d < G2C; ++d) o2[r + d] = o[d];
    }
}

// conv[l][g] = mean_n( tpg[l,n,g] * g2[l,n,g] * sw2[l,n] )
__global__ void k_conv(const float* __restrict__ tpg, const float* __restrict__ g2,
                       const float* __restrict__ sw2, float* __restrict__ conv)
{
    int idx = blockIdx.x * blockDim.x + threadIdx.x;
    if (idx >= NLOC * G2C) return;
    int l = idx >> 4, g = idx & 15;
    float acc = 0.0f;
    for (int n = 0; n < RF_NSEL; ++n) {
        size_t r = (size_t)l * RF_NSEL + n;
        acc = fmaf(tpg[r * G2C + g] * g2[r * G2C + g], sw2[r], acc);
    }
    conv[idx] = acc * (1.0f / RF_NSEL);
}

// Build lin1 input row: [g1 | conv | grrg(16x4) | drrd(128x4)]
__global__ __launch_bounds__(128)
void k_xin(const float* __restrict__ g1, const float* __restrict__ conv,
           const float* __restrict__ h2, const float* __restrict__ g2,
           const float* __restrict__ gg1, const float* __restrict__ sw2,
           float* __restrict__ xin)
{
    __shared__ float s_h2g2[3][G2C];
    __shared__ float s_h2g1[3][G1C];
    const int l = blockIdx.x, t = threadIdx.x;
    const size_t base = (size_t)l * RF_NSEL;
    if (t < 48) {
        int d = t / G2C, g = t % G2C;
        float acc = 0.0f;
        for (int n = 0; n < RF_NSEL; ++n)
            acc = fmaf(h2[(base + n) * 3 + d] * sw2[base + n], g2[(base + n) * G2C + g], acc);
        s_h2g2[d][g] = acc * (1.0f / RF_NSEL);
    }
    for (int e = t; e < 3 * G1C; e += 128) {
        int d = e / G1C, c = e % G1C;
        float acc = 0.0f;
        for (int n = 0; n < RF_NSEL; ++n)
            acc = fmaf(h2[(base + n) * 3 + d] * sw2[base + n], gg1[((base + n) << 7) + c], acc);
        s_h2g1[d][c] = acc * (1.0f / RF_NSEL);
    }
    __syncthreads();
    float* xr = xin + (size_t)l * DIN1;
    xr[t] = g1[(size_t)l * G1C + t];                                  // 0..127
    if (t < G2C) xr[G1C + t] = conv[l * G2C + t];                     // 128..143
    if (t < 64) {                                                     // 144..207
        int g = t >> 2, a = t & 3;
        float acc = 0.0f;
#pragma unroll
        for (int d = 0; d < 3; ++d) acc = fmaf(s_h2g2[d][g], s_h2g2[d][a], acc);
        xr[144 + t] = acc * (1.0f / 3.0f);
    }
    for (int e = t; e < 512; e += 128) {                              // 208..719
        int c = e >> 2, a = e & 3;
        float acc = 0.0f;
#pragma unroll
        for (int d = 0; d < 3; ++d) acc = fmaf(s_h2g1[d][c], s_h2g1[d][a], acc);
        xr[208 + e] = acc * (1.0f / 3.0f);
    }
}

// g1 neighbor attention: block = atom; 4 heads x 40-wide softmax
__global__ __launch_bounds__(128)
void k_attn1(const float* __restrict__ q1, const float* __restrict__ k1,
             const float* __restrict__ v1, const float* __restrict__ sw2,
             float* __restrict__ o1)
{
    __shared__ float s_aw[NH1][RF_NSEL];
    const int l = blockIdx.x, t = threadIdx.x;
    const size_t base = (size_t)l * RF_NSEL;
    for (int e = t; e < NH1 * RF_NSEL; e += 128) {
        int h = e / RF_NSEL, n = e % RF_NSEL;
        const float* q = q1 + (size_t)l * (NH1 * HD1) + h * HD1;
        const float* k = k1 + (base + n) * (NH1 * HD1) + h * HD1;
        float acc = 0.0f;
#pragma unroll 8
        for (int d = 0; d < HD1; ++d) acc = fmaf(q[d], k[d], acc);
        s_aw[h][n] = acc * 0.125f;                  // 1/sqrt(64)
    }
    __syncthreads();
    if (t < NH1) {
        float mx = -3.0e38f;
        for (int n = 0; n < RF_NSEL; ++n) mx = fmaxf(mx, s_aw[t][n]);
        float sum = 0.0f;
        for (int n = 0; n < RF_NSEL; ++n) { float e = __expf(s_aw[t][n] - mx); s_aw[t][n] = e; sum += e; }
        float inv = 1.0f / sum;
        for (int n = 0; n < RF_NSEL; ++n) s_aw[t][n] *= inv * sw2[base + n];
    }
    __syncthreads();
    for (int e = t; e < NH1 * G1C; e += 128) {
        int h = e >> 7, d = e & 127;
        float acc = 0.0f;
        for (int n = 0; n < RF_NSEL; ++n)
            acc = fmaf(s_aw[h][n], v1[(base + n) * (NH1 * G1C) + h * G1C + d], acc);
        o1[(size_t)l * (NH1 * G1C) + e] = acc;
    }
}

// g2_new = (g2 + g2_self + tanh(tgg)*swn + g2_attn) / 2
__global__ void k_g2_update(const float* __restrict__ g2, const float* __restrict__ g2s,
                            const float* __restrict__ tgg, const float* __restrict__ g2a,
                            const float* __restrict__ sw2, float* __restrict__ g2n)
{
    size_t idx = (size_t)blockIdx.x * blockDim.x + threadIdx.x;
    if (idx >= (size_t)RROWS * G2C) return;
    int r = (int)(idx >> 4);
    float g1g1 = tanhf(tgg[idx]) * sw2[r];
    g2n[idx] = (g2[idx] + g2s[idx] + g1g1 + g2a[idx]) * 0.5f;
}

// g1_new = (g1 + g1_mlp + g1_attn) / sqrt(3)
__global__ void k_g1_update(const float* __restrict__ g1, const float* __restrict__ g1m,
                            const float* __restrict__ g1a, float* __restrict__ g1n)
{
    int idx = blockIdx.x * blockDim.x + threadIdx.x;
    if (idx >= NLOC * G1C) return;
    g1n[idx] = (g1[idx] + g1m[idx] + g1a[idx]) * 0.57735026919f;
}

// out = concat([g1, tebd_ext[:NLOC]], -1)   -> (4096, 136) f32
__global__ void k_final(const float* __restrict__ g1, const float* __restrict__ tebd_ext,
                        float* __restrict__ out)
{
    int idx = blockIdx.x * blockDim.x + threadIdx.x;
    if (idx >= NLOC * (G1C + TEBD)) return;
    int l = idx / (G1C + TEBD), c = idx % (G1C + TEBD);
    out[idx] = (c < G1C) ? g1[(size_t)l * G1C + c] : tebd_ext[l * TEBD + (c - G1C)];
}

// ---------------------------------------------------------------------------
// Host orchestration.
// Input leaf order (recursive dict-insertion-order flatten of setup_inputs):
//  0 coord_ext  1 atype_ext  2 nlist_repinit  3 nlist_repformer  4 mapping
//  5..10  embed (w,b)x3   11,12 g1_trans   13,14 g2_embd   15 tebd
//  16+14*L: lin1_w lin1_b lin2_w lin2_b pg1g2_w g1g1_w a2_q a2_k a2_v a2_o
//           a1_q a1_k a1_v a1_o
// ---------------------------------------------------------------------------
extern "C" void kernel_launch(void* const* d_in, const int* in_sizes, int n_in,
                              void* d_out, int out_size, void* d_ws, size_t ws_size,
                              hipStream_t stream)
{
    (void)in_sizes; (void)n_in; (void)out_size; (void)ws_size;

    const float* coord    = (const float*)d_in[0];
    const int*   atype    = (const int*)d_in[1];
    const int*   nlist_ri = (const int*)d_in[2];
    const int*   nlist_rf = (const int*)d_in[3];
    const int*   mapping  = (const int*)d_in[4];
    const float* eW1 = (const float*)d_in[5];  const float* eB1 = (const float*)d_in[6];
    const float* eW2 = (const float*)d_in[7];  const float* eB2 = (const float*)d_in[8];
    const float* eW3 = (const float*)d_in[9];  const float* eB3 = (const float*)d_in[10];
    const float* g1tW = (const float*)d_in[11]; const float* g1tB = (const float*)d_in[12];
    const float* g2eW = (const float*)d_in[13]; const float* g2eB = (const float*)d_in[14];
    const float* tebd = (const float*)d_in[15];
    float* out = (float*)d_out;

    // workspace carve-up (floats)
    float* W = (float*)d_ws;
    size_t off = 0;
    auto take = [&](size_t n) { float* p = W + off; off += n; return p; };
    float* tebd_ext = take((size_t)NALL * TEBD);
    float* g1a   = take((size_t)NLOC * G1C);
    float* g1b   = take((size_t)NLOC * G1C);
    float* g2a   = take((size_t)RROWS * G2C);
    float* g2b   = take((size_t)RROWS * G2C);
    float* h2b   = take((size_t)RROWS * 3);
    float* sw2   = take((size_t)RROWS);
    float* res   = take((size_t)NLOC * 1600);
    float* gg1   = take((size_t)RROWS * G1C);
    float* g1gg1 = take((size_t)RROWS * G1C);
    float* k1b   = take((size_t)RROWS * NH1 * HD1);
    float* v1b   = take((size_t)RROWS * NH1 * G1C);
    float* q2b   = take((size_t)RROWS * NH2 * G2C);
    float* k2b   = take((size_t)RROWS * NH2 * G2C);
    float* v2b   = take((size_t)RROWS * NH2 * G2C);
    float* o2b   = take((size_t)RROWS * NH2 * G2C);
    float* tpg   = take((size_t)RROWS * G2C);
    float* tgg   = take((size_t)RROWS * G2C);
    float* g2s   = take((size_t)RROWS * G2C);
    float* g2at  = take((size_t)RROWS * G2C);
    float* convb = take((size_t)NLOC * G2C);
    float* xin   = take((size_t)NLOC * DIN1);
    float* q1b   = take((size_t)NLOC * NH1 * HD1);
    float* o1b   = take((size_t)NLOC * NH1 * G1C);
    float* g1mlp = take((size_t)NLOC * G1C);
    float* g1att = take((size_t)NLOC * G1C);

    // --- stage 0: type embedding gather ---
    k_tebd<<<(NALL * TEBD + 255) / 256, 256, 0, stream>>>(atype, tebd, tebd_ext);

    // --- stage 1: repinit -> res, then g1 = res @ g1_trans + b (WMMA) ---
    k_repinit<<<NLOC, 128, 0, stream>>>(coord, nlist_ri, tebd_ext,
                                        eW1, eB1, eW2, eB2, eW3, eB3, res);
    gemm(res, g1tW, g1tB, g1a, NLOC, G1C, 1600, 0, stream);

    // --- stage 2: repformer init ---
    k_rf_init<<<(RROWS + 255) / 256, 256, 0, stream>>>(coord, nlist_rf, g2eW, g2eB,
                                                       g2a, h2b, sw2);

    float* g1_cur = g1a; float* g1_nxt = g1b;
    float* g2_cur = g2a; float* g2_nxt = g2b;

    for (int L = 0; L < NLAYERS; ++L) {
        const int pb = 16 + L * 14;
        const float* lin1_w = (const float*)d_in[pb + 0];
        const float* lin1_b = (const float*)d_in[pb + 1];
        const float* lin2_w = (const float*)d_in[pb + 2];
        const float* lin2_b = (const float*)d_in[pb + 3];
        const float* pg1g2  = (const float*)d_in[pb + 4];
        const float* g1g1w  = (const float*)d_in[pb + 5];
        const float* a2_q   = (const float*)d_in[pb + 6];
        const float* a2_k   = (const float*)d_in[pb + 7];
        const float* a2_v   = (const float*)d_in[pb + 8];
        const float* a2_o   = (const float*)d_in[pb + 9];
        const float* a1_q   = (const float*)d_in[pb + 10];
        const float* a1_k   = (const float*)d_in[pb + 11];
        const float* a1_v   = (const float*)d_in[pb + 12];
        const float* a1_o   = (const float*)d_in[pb + 13];

        // neighbor gather of g1 (+ elementwise product for g1g1 path)
        {
            size_t tot = (size_t)RROWS * G1C;
            k_gather<<<(unsigned)((tot + 255) / 256), 256, 0, stream>>>(
                g1_cur, nlist_rf, mapping, gg1, g1gg1);
        }

        // g2-side projections (WMMA)
        gemm(g1gg1, g1g1w, nullptr, tgg, RROWS, G2C, G1C, 0, stream);
        gemm(gg1, pg1g2, nullptr, tpg, RROWS, G2C, G1C, 0, stream);
        gemm(g2_cur, lin2_w, lin2_b, g2s, RROWS, G2C, G2C, 1, stream);
        gemm(g2_cur, a2_q, nullptr, q2b, RROWS, NH2 * G2C, G2C, 0, stream);
        gemm(g2_cur, a2_k, nullptr, k2b, RROWS, NH2 * G2C, G2C, 0, stream);
        gemm(g2_cur, a2_v, nullptr, v2b, RROWS, NH2 * G2C, G2C, 0, stream);

        // g2 attention + output proj + update
        {
            dim3 g(NLOC, NH2);
            k_attn2<<<g, 64, 0, stream>>>(q2b, k2b, v2b, sw2, o2b);
        }
        gemm(o2b, a2_o, nullptr, g2at, RROWS, G2C, NH2 * G2C, 0, stream);
        k_g2_update<<<(RROWS * G2C + 255) / 256, 256, 0, stream>>>(
            g2_cur, g2s, tgg, g2at, sw2, g2_nxt);

        // g1-side reductions + projections (WMMA)
        k_conv<<<(NLOC * G2C + 255) / 256, 256, 0, stream>>>(tpg, g2_cur, sw2, convb);
        gemm(gg1, a1_k, nullptr, k1b, RROWS, NH1 * HD1, G1C, 0, stream);
        gemm(gg1, a1_v, nullptr, v1b, RROWS, NH1 * G1C, G1C, 0, stream);
        gemm(g1_cur, a1_q, nullptr, q1b, NLOC, NH1 * HD1, G1C, 0, stream);
        k_xin<<<NLOC, 128, 0, stream>>>(g1_cur, convb, h2b, g2_cur, gg1, sw2, xin);
        gemm(xin, lin1_w, lin1_b, g1mlp, NLOC, G1C, DIN1, 1, stream);

        k_attn1<<<NLOC, 128, 0, stream>>>(q1b, k1b, v1b, sw2, o1b);
        gemm(o1b, a1_o, nullptr, g1att, NLOC, G1C, NH1 * G1C, 0, stream);
        k_g1_update<<<(NLOC * G1C + 255) / 256, 256, 0, stream>>>(
            g1_cur, g1mlp, g1att, g1_nxt);

        // ping-pong
        float* t1 = g1_cur; g1_cur = g1_nxt; g1_nxt = t1;
        float* t2 = g2_cur; g2_cur = g2_nxt; g2_nxt = t2;
    }

    k_final<<<(NLOC * (G1C + TEBD) + 255) / 256, 256, 0, stream>>>(g1_cur, tebd_ext, out);
}
